// SingleStageFIRFilter_49409303773366
// MI455X (gfx1250) — matrix operand order repaired and checked by reading, not compiled, not run on hardware
//
#include <hip/hip_runtime.h>
#include <hip/hip_bf16.h>
#include <math.h>

#define FRAME_PERIOD 80
#define IR_LEN 2000
#define NFFT 4096
#define NFRM 300
#define NB 2
#define NSMP 24000
#define NK 2049          // NFFT/2 + 1 unique spectrum points (Hermitian)
#define KE 2064          // padded stride for Er/Ei per (b,frame)
#define HSTRIDE 2048     // padded stride for flipped IR per (b,frame)
#define KPAD 2016        // IR length padded to multiple of 32 (WMMA K-tile)

typedef __attribute__((ext_vector_type(16))) _Float16 v16h;
typedef __attribute__((ext_vector_type(8)))  float    v8f;

// ---------------------------------------------------------------------------
// Kernel 1: E[k] = exp( DFT_25(mc)[k] ) for k = 0..2048, per (b, frame).
// C[k] = sum_m mc[m] * e^{-2*pi*i*k*m/4096}; E = exp(Cr) * (cos Ci, sin Ci).
// ---------------------------------------------------------------------------
__global__ void spectrum_kernel(const float* __restrict__ mc,
                                float* __restrict__ Er,
                                float* __restrict__ Ei) {
  int idx = blockIdx.x * blockDim.x + threadIdx.x;
  int total = NB * NFRM * NK;
  if (idx >= total) return;
  int bn = idx / NK;
  int k  = idx - bn * NK;
  const float* c = mc + bn * 25;
  float w = (float)k * (6.283185307179586f / (float)NFFT);
  float cr = 0.f, ci = 0.f;
#pragma unroll
  for (int m = 0; m < 25; ++m) {
    float s, co;
    sincosf(w * (float)m, &s, &co);
    float cm = c[m];
    cr = fmaf(cm, co, cr);
    ci = fmaf(-cm, s, ci);
  }
  float e = expf(cr);
  float s, co;
  sincosf(ci, &s, &co);
  Er[bn * KE + k] = e * co;
  Ei[bn * KE + k] = e * s;
}

// ---------------------------------------------------------------------------
// Kernel 2: flipped min-phase IR via Hermitian inverse DFT.
// h[l] = (Er[0] + (-1)^l * Er[2048] + 2*sum_{k=1}^{2047}(Er cos - Ei sin))/4096
// stored flipped: HH[p] = h[1999 - p]; tail [2000,2048) zeroed for WMMA pad.
// Rotation recurrence (4 FMA/step) + exact sincos refresh every 256 steps.
// ---------------------------------------------------------------------------
__global__ void __launch_bounds__(256) ir_kernel(const float* __restrict__ Er,
                                                 const float* __restrict__ Ei,
                                                 float* __restrict__ HH) {
  __shared__ float sEr[NK];
  __shared__ float sEi[NK];
  int bn   = blockIdx.x >> 3;
  int tile = blockIdx.x & 7;
  for (int i = threadIdx.x; i < NK; i += 256) {
    sEr[i] = Er[bn * KE + i];
    sEi[i] = Ei[bn * KE + i];
  }
  __syncthreads();
  int p = tile * 256 + (int)threadIdx.x;       // flipped output index 0..2047
  if (p >= HSTRIDE) return;
  if (p >= IR_LEN) { HH[bn * HSTRIDE + p] = 0.f; return; }  // zero pad tail
  int l = (IR_LEN - 1) - p;                    // original time index
  float delta = (float)l * (6.283185307179586f / (float)NFFT);
  float cd, sd;
  sincosf(delta, &sd, &cd);
  float c = cd, s = sd;                        // theta(k=1) = delta
  float acc = 0.f;
  for (int k = 1; k < NFFT / 2; ++k) {
    if ((k & 255) == 0) {                      // drift refresh
      sincosf(delta * (float)k, &s, &c);
    }
    acc = fmaf(sEr[k], c, acc);
    acc = fmaf(-sEi[k], s, acc);
    float cn = fmaf(c, cd, -s * sd);
    s = fmaf(s, cd, c * sd);
    c = cn;
  }
  float sign = (l & 1) ? -1.f : 1.f;
  float h = (sEr[0] + sign * sEr[NFFT / 2] + 2.f * acc) * (1.0f / (float)NFFT);
  HH[bn * HSTRIDE + p] = h;
}

// ---------------------------------------------------------------------------
// Kernel 3: time-varying FIR via WMMA.
// One wave per (b, frame, 16-sample tile). D = A x B + C with
//   A[m][k] = x_pad[t0 + m + k]           (Toeplitz window, f16)
//   B[k][0] = h_n[k], B[k][1] = h_{n+1}[k], cols 2..15 = 0
// accumulate f32 over K = 2016 (63 x v_wmma_f32_16x16x32_f16), then blend
//   y[t0+m] = (1-w)*D[m][0] + w*D[m][1],  w = ((t0+m) % 80)/80.
// B-fragment is built branch-free: per-lane LDS base pointer selected ONCE
// (h0 / h1 / zero array) so the K-loop is pure ds_load_b128 + v_wmma with
// EXEC all-1s throughout (no saveexec chains).
// ---------------------------------------------------------------------------
__global__ void __launch_bounds__(32) fir_wmma_kernel(const float* __restrict__ x,
                                                      const float* __restrict__ HH,
                                                      float* __restrict__ y) {
  __shared__ __align__(16) _Float16 xw[2048];
  __shared__ __align__(16) _Float16 h0[KPAD];
  __shared__ __align__(16) _Float16 h1[KPAD];
  __shared__ __align__(16) _Float16 hz[KPAD];   // all-zero column source
  __shared__ float sD[16][17];

  int bid  = blockIdx.x;
  int mt   = bid % 5;            // 16-sample tile within frame (80 = 5*16)
  int pair = bid / 5;
  int b    = pair / NFRM;
  int n    = pair % NFRM;
  int t0   = n * FRAME_PERIOD + mt * 16;

  int lane = threadIdx.x;
  const float* xb = x + b * NSMP;
  // Stage sliding window x_pad[t0 .. t0+2047] (left pad of 1999 zeros).
  for (int i = lane; i < 2048; i += 32) {
    int g = t0 + i - (IR_LEN - 1);
    float v = (g >= 0 && g < NSMP) ? xb[g] : 0.f;
    xw[i] = (_Float16)v;
  }
  const float* H0 = HH + (size_t)(b * NFRM + n) * HSTRIDE;
  int n1 = (n + 1 < NFRM) ? (n + 1) : (NFRM - 1);
  const float* H1 = HH + (size_t)(b * NFRM + n1) * HSTRIDE;
  for (int i = lane; i < KPAD; i += 32) {
    h0[i] = (_Float16)H0[i];
    h1[i] = (_Float16)H1[i];
    hz[i] = (_Float16)0.f;
  }
  __syncthreads();

  int mrow = lane & 15;
  int hi   = lane >> 4;
  // One-time per-lane column source select: col 0 -> h_n, col 1 -> h_{n+1},
  // cols 2..15 -> zeros.  Stays an LDS (addrspace 3) pointer -> ds loads.
  const _Float16* hsel = (mrow == 0) ? h0 : ((mrow == 1) ? h1 : hz);

  v8f acc = {0.f, 0.f, 0.f, 0.f, 0.f, 0.f, 0.f, 0.f};

  for (int kt = 0; kt < KPAD / 32; ++kt) {
    int kb = kt * 32;
    v16h a, bm;
    // A fragment (16-bit A 16x32): koff = j + 16*hi (matches B)
#pragma unroll
    for (int j = 0; j < 16; ++j) {
      int koff = j + 16 * hi;
      a[j] = xw[mrow + kb + koff];
    }
    // B fragment (16-bit B 32x16): lanes 0-15 hold K=kb..kb+15, lanes 16-31
    // hold K=kb+16..kb+31; contiguous 16 halves -> 2x ds_load_b128.
    int kof = kb + 16 * hi;
#pragma unroll
    for (int j = 0; j < 16; ++j) {
      bm[j] = hsel[kof + j];
    }
    acc = __builtin_amdgcn_wmma_f32_16x16x32_f16(
        /*neg_a=*/false, a, /*neg_b=*/false, bm,
        /*c_mod=*/(short)0, acc, /*reuse_a=*/false, /*reuse_b=*/false);
  }

  // D layout (f32 16x16): lane holds column N = lane&15, rows r + 8*hi.
#pragma unroll
  for (int r = 0; r < 8; ++r) {
    sD[r + 8 * hi][mrow] = acc[r];
  }
  __syncthreads();
  if (lane < 16) {
    int m = lane;
    float w = (float)(mt * 16 + m) * (1.0f / (float)FRAME_PERIOD);
    float out = (1.f - w) * sD[m][0] + w * sD[m][1];
    y[b * NSMP + t0 + m] = out;
  }
}

// ---------------------------------------------------------------------------
extern "C" void kernel_launch(void* const* d_in, const int* in_sizes, int n_in,
                              void* d_out, int out_size, void* d_ws, size_t ws_size,
                              hipStream_t stream) {
  const float* x  = (const float*)d_in[0];   // (2, 24000) f32
  const float* mc = (const float*)d_in[1];   // (2, 300, 25) f32
  float* y  = (float*)d_out;                 // (2, 24000) f32
  float* ws = (float*)d_ws;

  float* Er = ws;
  float* Ei = ws + (size_t)NB * NFRM * KE;
  float* HH = ws + (size_t)2 * NB * NFRM * KE;

  int total1 = NB * NFRM * NK;
  spectrum_kernel<<<(total1 + 255) / 256, 256, 0, stream>>>(mc, Er, Ei);
  ir_kernel<<<NB * NFRM * 8, 256, 0, stream>>>(Er, Ei, HH);
  fir_wmma_kernel<<<NB * NFRM * 5, 32, 0, stream>>>(x, HH, y);
}
